// SREL_intra_rep_25915832664321
// MI455X (gfx1250) — compile-verified, matrix operand order checked by reading, not run on hardware
//
#include <hip/hip_runtime.h>

// ---------------- problem constants (from reference) ----------------
#define BATCH   64
#define LS      1024
#define LW      1024
#define LY      2048
#define NSTEP   10
#define HIDDEN  4096
#define DIN     6144   // 2*LS + 2*LW + LY
#define KSPLIT  4      // K-dimension split for GEMM parallelism

typedef __bf16 v16bf __attribute__((ext_vector_type(16)));
typedef float  v8f   __attribute__((ext_vector_type(8)));

// K index inside a 32-wide K-step for 16-bit WMMA operands (ISA 7.12.2):
// lanes 0-15 hold K in {0..7} u {16..23}; lanes 16-31 hold K in {8..15} u {24..31}
__device__ __forceinline__ int kmap(int l, int e) {
    return (e < 8 ? e : e + 8) + ((l & 16) ? 8 : 0);
}

#define WMMA_BF16(A, Bm, C) \
    __builtin_amdgcn_wmma_f32_16x16x32_bf16(false, (A), false, (Bm), (short)0, (C), false, false)

// ---------------- weight pack: fp32 row-major [K,N] -> bf16 WMMA-B layout ----
__global__ void pack_w_kernel(const float* __restrict__ W, __bf16* __restrict__ dst,
                              int K, int N) {
    long d = (long)blockIdx.x * blockDim.x + threadIdx.x;
    long total = (long)K * N;
    if (d >= total) return;
    int e = (int)(d & 15);
    int l = (int)((d >> 4) & 31);
    long rest = d >> 9;
    int S = K >> 5;
    int s = (int)(rest % S);
    int t = (int)(rest / S);
    int n = t * 16 + (l & 15);
    int k = s * 32 + kmap(l, e);
    dst[d] = (__bf16)W[(long)k * N + n];
}

// ---------------- activation pack: fp32 row-major [M,K] -> bf16 WMMA-A layout -
__global__ void pack_a_kernel(const float* __restrict__ A, __bf16* __restrict__ dst,
                              int M, int K) {
    long d = (long)blockIdx.x * blockDim.x + threadIdx.x;
    long total = (long)M * K;
    if (d >= total) return;
    int e = (int)(d & 15);
    int l = (int)((d >> 4) & 31);
    long rest = d >> 9;
    int S = K >> 5;
    int s = (int)(rest % S);
    int mt = (int)(rest / S);
    int m = mt * 16 + (l & 15);
    int k = s * 32 + kmap(l, e);
    dst[d] = (__bf16)A[(long)m * K + k];
}

// ---------------- WMMA GEMM, M-blocked + K-split, 2-deep SW pipeline ----------
// K is a template parameter so all stream strides are compile-time constants
// (immediate offsets instead of 64-bit address recomputation in the hot loop).
// One wave32 computes the full M=64 column of one 16-wide N tile over K/KSPLIT.
// Per K-step: B loaded once, reused by 4 WMMAs into 4 independent accumulators.
// Operands are kept 2 K-steps ahead in registers to cover L2 latency.
// Writes fp32 partials: part[ks][64][N].
template<int K>
__global__ void gemm_wmma_kernel(const __bf16* __restrict__ Ap,
                                 const __bf16* __restrict__ Bp,
                                 float* __restrict__ part,
                                 int N) {
    constexpr int S    = K >> 5;        // K-steps total
    constexpr int Sper = S / KSPLIT;    // K-steps per wave (48 or 32 -> even)
    constexpr size_t AS = (size_t)S * 32;  // v16bf stride between A M-tiles

    int lane = threadIdx.x & 31;
    int wave = threadIdx.x >> 5;
    int wg   = blockIdx.x * (blockDim.x >> 5) + wave;
    int ntn  = N >> 4;
    int nt   = wg % ntn;
    int ks   = wg / ntn;
    if (ks >= KSPLIT) return;           // wave-uniform: EXEC stays all-1s for WMMA

    int s0 = ks * Sper;
    const v16bf* bP = (const v16bf*)(Bp + (((size_t)nt * S + s0) * 32 + lane) * 16);
    const v16bf* aP = (const v16bf*)(Ap + (((size_t)s0 * 32 + lane)) * 16);

    v8f acc0 = {}, acc1 = {}, acc2 = {}, acc3 = {};

    // stage 0 (s) and stage 1 (s+1) operands
    v16bf b_c  = bP[0];
    v16bf a0_c = aP[0 * AS], a1_c = aP[1 * AS], a2_c = aP[2 * AS], a3_c = aP[3 * AS];
    v16bf b_n  = bP[32];
    v16bf a0_n = aP[0 * AS + 32], a1_n = aP[1 * AS + 32],
          a2_n = aP[2 * AS + 32], a3_n = aP[3 * AS + 32];

    int s = 0;
#pragma unroll 1
    for (; s + 2 < Sper; s += 2) {
        size_t o2 = (size_t)(s + 2) * 32;
        size_t o3 = (size_t)(s + 3) * 32;
        // speculative prefetch ~8 K-steps ahead (lands inside our ws even past end)
        __builtin_prefetch(bP + (size_t)(s + 8) * 32, 0, 3);
        __builtin_prefetch(aP + 0 * AS + (size_t)(s + 8) * 32, 0, 3);
        __builtin_prefetch(aP + 1 * AS + (size_t)(s + 8) * 32, 0, 3);
        __builtin_prefetch(aP + 2 * AS + (size_t)(s + 8) * 32, 0, 3);
        __builtin_prefetch(aP + 3 * AS + (size_t)(s + 8) * 32, 0, 3);

        // issue loads for s+2, consume stage s
        v16bf b2  = bP[o2];
        v16bf a02 = aP[0 * AS + o2], a12 = aP[1 * AS + o2],
              a22 = aP[2 * AS + o2], a32 = aP[3 * AS + o2];
        acc0 = WMMA_BF16(a0_c, b_c, acc0);
        acc1 = WMMA_BF16(a1_c, b_c, acc1);
        acc2 = WMMA_BF16(a2_c, b_c, acc2);
        acc3 = WMMA_BF16(a3_c, b_c, acc3);

        // issue loads for s+3, consume stage s+1
        v16bf b3  = bP[o3];
        v16bf a03 = aP[0 * AS + o3], a13 = aP[1 * AS + o3],
              a23 = aP[2 * AS + o3], a33 = aP[3 * AS + o3];
        acc0 = WMMA_BF16(a0_n, b_n, acc0);
        acc1 = WMMA_BF16(a1_n, b_n, acc1);
        acc2 = WMMA_BF16(a2_n, b_n, acc2);
        acc3 = WMMA_BF16(a3_n, b_n, acc3);

        b_c = b2; a0_c = a02; a1_c = a12; a2_c = a22; a3_c = a32;
        b_n = b3; a0_n = a03; a1_n = a13; a2_n = a23; a3_n = a33;
    }
    // drain the last two stages (Sper is even)
    acc0 = WMMA_BF16(a0_c, b_c, acc0);
    acc1 = WMMA_BF16(a1_c, b_c, acc1);
    acc2 = WMMA_BF16(a2_c, b_c, acc2);
    acc3 = WMMA_BF16(a3_c, b_c, acc3);
    acc0 = WMMA_BF16(a0_n, b_n, acc0);
    acc1 = WMMA_BF16(a1_n, b_n, acc1);
    acc2 = WMMA_BF16(a2_n, b_n, acc2);
    acc3 = WMMA_BF16(a3_n, b_n, acc3);

    // C/D layout: VGPR r, lane: M = mt*16 + r + (lane[4]?8:0), N = nt*16 + lane%16
    int n    = nt * 16 + (lane & 15);
    int mofs = (lane & 16) ? 8 : 0;
    float* out = part + (size_t)ks * BATCH * N;
#pragma unroll
    for (int r = 0; r < 8; ++r) {
        out[(size_t)( 0 + mofs + r) * N + n] = acc0[r];
        out[(size_t)(16 + mofs + r) * N + n] = acc1[r];
        out[(size_t)(32 + mofs + r) * N + n] = acc2[r];
        out[(size_t)(48 + mofs + r) * N + n] = acc3[r];
    }
}

// ---------------- sum K-split partials + bias (+relu) -------------------------
__global__ void reduce_bias_kernel(const float* __restrict__ part,
                                   const float* __restrict__ bias,
                                   float* __restrict__ C,
                                   int MN, int N, int relu) {
    int idx = blockIdx.x * blockDim.x + threadIdx.x;
    if (idx >= MN) return;
    float v = 0.f;
#pragma unroll
    for (int k = 0; k < KSPLIT; ++k) v += part[(size_t)k * MN + idx];
    v += bias[idx % N];
    if (relu) v = fmaxf(v, 0.f);
    C[idx] = v;
}

// ---------------- per-step: s = exp(i*phi); fill x head + s_stack output -----
__global__ void build_x_kernel(const float* __restrict__ phi, float* __restrict__ x,
                               float* __restrict__ outS, int step) {
    int idx = blockIdx.x * blockDim.x + threadIdx.x;     // over BATCH*LS
    if (idx >= BATCH * LS) return;
    int b = idx >> 10, i = idx & (LS - 1);
    float p = phi[idx];
    float sn, cs;
    __sincosf(p, &sn, &cs);
    x[(size_t)b * DIN + i]      = cs;
    x[(size_t)b * DIN + LS + i] = sn;
    size_t base = ((size_t)(b * (NSTEP + 1) + step) * LS + i) * 2;  // complex64 interleave
    outS[base]     = cs;
    outS[base + 1] = sn;
}

// ---------------- one-time: constant tail of x, phi working copy -------------
__global__ void init_tail_kernel(const float* __restrict__ wr, const float* __restrict__ wi,
                                 const float* __restrict__ y, float* __restrict__ x,
                                 const float* __restrict__ phi_in, float* __restrict__ phi_ws) {
    int idx = blockIdx.x * blockDim.x + threadIdx.x;     // over BATCH*(DIN-2*LS)
    if (idx >= BATCH * (DIN - 2 * LS)) return;
    int b = idx / (DIN - 2 * LS);
    int c = idx % (DIN - 2 * LS);
    float v;
    if      (c < LW)        v = wr[b * LW + c];
    else if (c < 2 * LW)    v = wi[b * LW + (c - LW)];
    else                    v = y[c - 2 * LW];
    x[(size_t)b * DIN + 2 * LS + c] = v;
    if (idx < BATCH * LS) phi_ws[idx] = phi_in[idx];
}

// ---------------- beta = h_rho @ W2_rho + b2_rho (fp32 GEMV reduce) ----------
__global__ void gemv_rho_kernel(const float* __restrict__ h, const float* __restrict__ w2,
                                const float* __restrict__ b2, float* __restrict__ beta) {
    __shared__ float red[256];
    int b = blockIdx.x;
    float p = 0.f;
    for (int k = threadIdx.x; k < HIDDEN; k += 256)
        p += h[(size_t)b * HIDDEN + k] * w2[k];
    red[threadIdx.x] = p;
    __syncthreads();
    for (int ofs = 128; ofs > 0; ofs >>= 1) {
        if (threadIdx.x < ofs) red[threadIdx.x] += red[threadIdx.x + ofs];
        __syncthreads();
    }
    if (threadIdx.x == 0) beta[b] = red[0] + b2[0];
}

// ---------------- phi -= beta*eta; eta_stack output --------------------------
__global__ void update_kernel(const float* __restrict__ eta, const float* __restrict__ beta,
                              float* __restrict__ phi, float* __restrict__ outEta, int step) {
    int idx = blockIdx.x * blockDim.x + threadIdx.x;     // over BATCH*LS
    if (idx >= BATCH * LS) return;
    int b = idx >> 10, i = idx & (LS - 1);
    float e = eta[idx];
    outEta[((size_t)(b * NSTEP + step)) * LS + i] = e;
    phi[idx] -= beta[b] * e;
}

// =============================================================================
extern "C" void kernel_launch(void* const* d_in, const int* in_sizes, int n_in,
                              void* d_out, int out_size, void* d_ws, size_t ws_size,
                              hipStream_t stream) {
    const float* phi_in = (const float*)d_in[0];
    const float* w_real = (const float*)d_in[1];
    const float* w_imag = (const float*)d_in[2];
    const float* y      = (const float*)d_in[3];
    const float* W1e    = (const float*)d_in[4];
    const float* b1e    = (const float*)d_in[5];
    const float* W2e    = (const float*)d_in[6];
    const float* b2e    = (const float*)d_in[7];
    const float* W1r    = (const float*)d_in[8];
    const float* b1r    = (const float*)d_in[9];
    const float* W2r    = (const float*)d_in[10];
    const float* b2r    = (const float*)d_in[11];

    // workspace carve-up (~119 MB)
    char* ws = (char*)d_ws;
    size_t off = 0;
    auto carve = [&](size_t bytes) -> char* {
        char* p = ws + off;
        off = (off + bytes + 255) & ~(size_t)255;
        return p;
    };
    __bf16* W1e_p  = (__bf16*)carve((size_t)DIN * HIDDEN * 2);
    __bf16* W1r_p  = (__bf16*)carve((size_t)DIN * HIDDEN * 2);
    __bf16* W2e_p  = (__bf16*)carve((size_t)HIDDEN * LS * 2);
    float*  x_f32  = (float*) carve((size_t)BATCH * DIN * 4);
    __bf16* x_p    = (__bf16*)carve((size_t)BATCH * DIN * 2);
    float*  h_eta  = (float*) carve((size_t)BATCH * HIDDEN * 4);
    float*  h_rho  = (float*) carve((size_t)BATCH * HIDDEN * 4);
    __bf16* h_p    = (__bf16*)carve((size_t)BATCH * HIDDEN * 2);
    float*  eta_f  = (float*) carve((size_t)BATCH * LS * 4);
    float*  beta_f = (float*) carve((size_t)BATCH * 4);
    float*  phi_ws = (float*) carve((size_t)BATCH * LS * 4);
    float*  part   = (float*) carve((size_t)KSPLIT * BATCH * HIDDEN * 4);
    (void)ws_size; (void)in_sizes; (void)n_in; (void)out_size;

    float* outS   = (float*)d_out;                                 // complex64 -> 2 floats
    float* outEta = outS + (size_t)BATCH * (NSTEP + 1) * LS * 2;

    // ---- one-time: bf16 weight repack (after this, ~109MB of weights live in L2)
    {
        long t1 = (long)DIN * HIDDEN;
        pack_w_kernel<<<(int)((t1 + 255) / 256), 256, 0, stream>>>(W1e, W1e_p, DIN, HIDDEN);
        pack_w_kernel<<<(int)((t1 + 255) / 256), 256, 0, stream>>>(W1r, W1r_p, DIN, HIDDEN);
        long t2 = (long)HIDDEN * LS;
        pack_w_kernel<<<(int)((t2 + 255) / 256), 256, 0, stream>>>(W2e, W2e_p, HIDDEN, LS);
        init_tail_kernel<<<(BATCH * (DIN - 2 * LS) + 255) / 256, 256, 0, stream>>>(
            w_real, w_imag, y, x_f32, phi_in, phi_ws);
    }

    const int g_el = (BATCH * LS) / 256;                      // elementwise grids
    const int g_g1 = ((HIDDEN / 16) * KSPLIT) / 8;            // 128 blocks (1024 waves)
    const int g_g2 = ((LS / 16) * KSPLIT) / 8;                // 32 blocks  (256 waves)
    const int g_r1 = (BATCH * HIDDEN) / 256;
    const int g_r2 = (BATCH * LS) / 256;

    for (int step = 0; step < NSTEP; ++step) {
        build_x_kernel<<<g_el, 256, 0, stream>>>(phi_ws, x_f32, outS, step);
        pack_a_kernel<<<(BATCH * DIN) / 256, 256, 0, stream>>>(x_f32, x_p, BATCH, DIN);

        gemm_wmma_kernel<DIN><<<g_g1, 256, 0, stream>>>(x_p, W1e_p, part, HIDDEN);
        reduce_bias_kernel<<<g_r1, 256, 0, stream>>>(part, b1e, h_eta, BATCH * HIDDEN, HIDDEN, 1);
        gemm_wmma_kernel<DIN><<<g_g1, 256, 0, stream>>>(x_p, W1r_p, part, HIDDEN);
        reduce_bias_kernel<<<g_r1, 256, 0, stream>>>(part, b1r, h_rho, BATCH * HIDDEN, HIDDEN, 1);

        pack_a_kernel<<<(BATCH * HIDDEN) / 256, 256, 0, stream>>>(h_eta, h_p, BATCH, HIDDEN);
        gemm_wmma_kernel<HIDDEN><<<g_g2, 256, 0, stream>>>(h_p, W2e_p, part, LS);
        reduce_bias_kernel<<<g_r2, 256, 0, stream>>>(part, b2e, eta_f, BATCH * LS, LS, 0);

        gemv_rho_kernel<<<BATCH, 256, 0, stream>>>(h_rho, W2r, b2r, beta_f);
        update_kernel<<<g_el, 256, 0, stream>>>(eta_f, beta_f, phi_ws, outEta, step);
    }
    // final s_stack slice (step == NSTEP) from phi_N
    build_x_kernel<<<g_el, 256, 0, stream>>>(phi_ws, x_f32, outS, NSTEP);
}
